// gate_20091857011522
// MI455X (gfx1250) — matrix operand order repaired and checked by reading, not compiled, not run on hardware
//
#include <hip/hip_runtime.h>

// Problem constants (reference: B=8, T=2048, D=1024, E=8, top-k=2)
#define BT 16384   // B*T tokens
#define DD 1024    // feature dim
#define EE 8       // experts

typedef __bf16 bf16;
typedef __attribute__((ext_vector_type(16))) __bf16 v16bf;
typedef __attribute__((ext_vector_type(8)))  __bf16 v8bf;
typedef __attribute__((ext_vector_type(8)))  float  v8f;

// ---------------------------------------------------------------------------
// Kernel 1: f32 -> bf16 conversion (grid-stride). Bandwidth-trivial vs GEMM.
// ---------------------------------------------------------------------------
__global__ void cvt_bf16_kernel(const float* __restrict__ src,
                                bf16* __restrict__ dst, int n) {
  int i = blockIdx.x * blockDim.x + threadIdx.x;
  int stride = gridDim.x * blockDim.x;
  for (; i < n; i += stride) dst[i] = (bf16)src[i];
}

// ---------------------------------------------------------------------------
// Kernel 2: routing.  One wave32 per token computes 16 dot products
// (x . Wt[e], x . Wf[e]), then top-2 softmax + dense softmax + alpha blend.
// Writes mix[BT, E] (f32).  ~0.5 GFLOP total -> negligible.
// ---------------------------------------------------------------------------
__global__ void mix_kernel(const float* __restrict__ x,
                           const float* __restrict__ Wt, const float* __restrict__ btb,
                           const float* __restrict__ Wf, const float* __restrict__ bfb,
                           const float* __restrict__ alpha,
                           float* __restrict__ mix) {
  const int wid  = threadIdx.x >> 5;
  const int lane = threadIdx.x & 31;
  const int tok  = blockIdx.x * (blockDim.x >> 5) + wid;
  const float* xr = x + (size_t)tok * DD;

  float at[EE], af[EE];
#pragma unroll
  for (int e = 0; e < EE; ++e) { at[e] = 0.f; af[e] = 0.f; }

  for (int d = lane; d < DD; d += 32) {
    float xv = xr[d];
#pragma unroll
    for (int e = 0; e < EE; ++e) {
      at[e] += xv * Wt[e * DD + d];
      af[e] += xv * Wf[e * DD + d];
    }
  }
  // wave32 butterfly reduction
#pragma unroll
  for (int e = 0; e < EE; ++e) {
#pragma unroll
    for (int off = 16; off >= 1; off >>= 1) {
      at[e] += __shfl_xor(at[e], off, 32);
      af[e] += __shfl_xor(af[e], off, 32);
    }
  }
  if (lane == 0) {
    float lt[EE], lf[EE];
#pragma unroll
    for (int e = 0; e < EE; ++e) { lt[e] = at[e] + btb[e]; lf[e] = af[e] + bfb[e]; }

    // top-2 over lt (track values, no dynamic register indexing)
    int i1 = 0; float v1 = lt[0];
#pragma unroll
    for (int e = 1; e < EE; ++e) if (lt[e] > v1) { v1 = lt[e]; i1 = e; }
    int i2 = -1; float v2 = -3.4e38f;
#pragma unroll
    for (int e = 0; e < EE; ++e) if (e != i1 && lt[e] > v2) { v2 = lt[e]; i2 = e; }

    // softmax over the two selected logits
    float e2  = __expf(v2 - v1);
    float inv = 1.f / (1.f + e2);
    float tp1 = inv, tp2 = e2 * inv;

    // dense softmax over lf
    float m = lf[0];
#pragma unroll
    for (int e = 1; e < EE; ++e) m = fmaxf(m, lf[e]);
    float fp[EE]; float s = 0.f;
#pragma unroll
    for (int e = 0; e < EE; ++e) { fp[e] = __expf(lf[e] - m); s += fp[e]; }
    float rs = 1.f / s;

    float a = 1.f / (1.f + __expf(-alpha[0]));
#pragma unroll
    for (int e = 0; e < EE; ++e) {
      float tp = (e == i1) ? tp1 : ((e == i2) ? tp2 : 0.f);
      mix[(size_t)tok * EE + e] = a * tp + (1.f - a) * fp[e] * rs;
    }
  }
}

// ---------------------------------------------------------------------------
// Kernel 3: the 275-GFLOP fused GEMM.
//   out[t,o] = sum_e sigmoid( (x @ Wg[e]^T)[t,o] + bg[e,o] ) * mix[t,e]
// 128x128 tile per workgroup, 8 waves (2x4), 64x32 per wave = 4x2 WMMA tiles.
// bf16 A/B, f32 accumulate via v_wmma_f32_16x16x32_bf16; expert loop fused so
// the [B,T,D,E] intermediate never touches memory.
// ---------------------------------------------------------------------------
__global__ __launch_bounds__(256)
void moe_gate_kernel(const bf16* __restrict__ xh, const bf16* __restrict__ wgh,
                     const float* __restrict__ bg, const float* __restrict__ mix,
                     float* __restrict__ out) {
  __shared__ float s_mix[128 * EE];  // mix rows for this tile  [row][e]
  __shared__ float s_bg[EE * 128];   // bias cols for this tile [e][col]

  const int tid  = threadIdx.x;
  const int lane = tid & 31;
  const int wid  = tid >> 5;
  const int wm = wid >> 2;   // 0..1  (M waves)
  const int wn = wid & 3;    // 0..3  (N waves)
  const int gRow0 = blockIdx.y * 128;
  const int gCol0 = blockIdx.x * 128;

  for (int i = tid; i < 128 * EE; i += 256)
    s_mix[i] = mix[(size_t)gRow0 * EE + i];               // contiguous slice
  for (int i = tid; i < EE * 128; i += 256)
    s_bg[i] = bg[(size_t)(i >> 7) * DD + gCol0 + (i & 127)];
  __syncthreads();

  const int lm = lane & 15;   // position within 16 (M row for A, N col for B/C)
  const int lh = lane >> 4;   // lane half
  const int rowW = wm * 64;
  const int colW = wn * 32;

  const v8f zeroV = {};
  v8f acc[4][2];              // cross-expert accumulator
#pragma unroll
  for (int ms = 0; ms < 4; ++ms)
#pragma unroll
    for (int ns = 0; ns < 2; ++ns) acc[ms][ns] = zeroV;

  for (int e = 0; e < EE; ++e) {
    const bf16* We = wgh + (size_t)e * DD * DD;

    v8f c[4][2];              // per-expert GEMM accumulator
#pragma unroll
    for (int ms = 0; ms < 4; ++ms)
#pragma unroll
      for (int ns = 0; ns < 2; ++ns) c[ms][ns] = zeroV;

    for (int k0 = 0; k0 < DD; k0 += 32) {
      v16bf a[4], b[2];
      // A fragments: 16-bit A 16x32 layout — lane holds row lm, K-chunks
      // [lh*8, +8) and [lh*8+16, +8)  -> two 16B loads
      const int ka = k0 + (lh << 3);
#pragma unroll
      for (int ms = 0; ms < 4; ++ms) {
        const bf16* pa = xh + (size_t)(gRow0 + rowW + ms * 16 + lm) * DD + ka;
        v8bf lo = *(const v8bf*)pa;
        v8bf hi = *(const v8bf*)(pa + 16);
        a[ms] = __builtin_shufflevector(lo, hi,
                 0,1,2,3,4,5,6,7,8,9,10,11,12,13,14,15);
      }
      // B fragments: 16-bit B 32x16 layout — lane holds col lm, K-chunks
      // [lh*8, +8) and [lh*8+16, +8); B col n of (x . Wg^T) = Wg row n (contiguous)
      const int kb = k0 + (lh << 3);
#pragma unroll
      for (int ns = 0; ns < 2; ++ns) {
        const bf16* pb = We + (size_t)(gCol0 + colW + ns * 16 + lm) * DD + kb;
        v8bf lo = *(const v8bf*)pb;
        v8bf hi = *(const v8bf*)(pb + 16);
        b[ns] = __builtin_shufflevector(lo, hi,
                 0,1,2,3,4,5,6,7,8,9,10,11,12,13,14,15);
      }
#pragma unroll
      for (int ms = 0; ms < 4; ++ms)
#pragma unroll
        for (int ns = 0; ns < 2; ++ns)
          c[ms][ns] = __builtin_amdgcn_wmma_f32_16x16x32_bf16(
              false, a[ms], false, b[ns], (short)0, c[ms][ns], false, false);
    }

    // fused epilogue: sigmoid(gemm + bias) * mix, summed over experts
#pragma unroll
    for (int ms = 0; ms < 4; ++ms) {
#pragma unroll
      for (int ns = 0; ns < 2; ++ns) {
        const int lc = colW + ns * 16 + lm;
        const float bgv = s_bg[e * 128 + lc];
#pragma unroll
        for (int v = 0; v < 8; ++v) {
          // C/D layout: lane half selects M+8; VGPR v is M row v
          const int lr = rowW + ms * 16 + lh * 8 + v;
          float z = c[ms][ns][v] + bgv;
          float g = 1.f / (1.f + __expf(-z));
          acc[ms][ns][v] += g * s_mix[lr * EE + e];
        }
      }
    }
  }

  // store f32 output tile
#pragma unroll
  for (int ms = 0; ms < 4; ++ms) {
#pragma unroll
    for (int ns = 0; ns < 2; ++ns) {
      const int col = gCol0 + colW + ns * 16 + lm;
#pragma unroll
      for (int v = 0; v < 8; ++v) {
        const int row = gRow0 + rowW + ms * 16 + lh * 8 + v;
        out[(size_t)row * DD + col] = acc[ms][ns][v];
      }
    }
  }
}

// ---------------------------------------------------------------------------
// Launch: inputs per setup_inputs order:
// 0:x 1:Wg 2:bg 3:Wf 4:bf 5:Wt 6:bt 7:alpha 8:num
// ---------------------------------------------------------------------------
extern "C" void kernel_launch(void* const* d_in, const int* in_sizes, int n_in,
                              void* d_out, int out_size, void* d_ws, size_t ws_size,
                              hipStream_t stream) {
  const float* x     = (const float*)d_in[0];
  const float* Wg    = (const float*)d_in[1];
  const float* bg    = (const float*)d_in[2];
  const float* Wf    = (const float*)d_in[3];
  const float* bfp   = (const float*)d_in[4];
  const float* Wt    = (const float*)d_in[5];
  const float* btp   = (const float*)d_in[6];
  const float* alpha = (const float*)d_in[7];
  float* out = (float*)d_out;

  // workspace layout (bytes):
  //   [0, 32MB)        x in bf16           (BT*DD elems)
  //   [32MB, 48MB)     Wg in bf16          (EE*DD*DD elems)
  //   [48MB, ...)  mix f32             (BT*EE elems)
  char* ws = (char*)d_ws;
  bf16*  xh   = (bf16*)ws;
  bf16*  wgh  = (bf16*)(ws + (size_t)BT * DD * 2);
  float* mixb = (float*)(ws + (size_t)BT * DD * 2 + (size_t)EE * DD * DD * 2);

  cvt_bf16_kernel<<<2048, 256, 0, stream>>>(x,  xh,  BT * DD);
  cvt_bf16_kernel<<<2048, 256, 0, stream>>>(Wg, wgh, EE * DD * DD);
  mix_kernel<<<BT / 8, 256, 0, stream>>>(x, Wt, btp, Wf, bfp, alpha, mixb);

  dim3 grid(DD / 128, BT / 128);   // 8 x 128 = 1024 workgroups
  moe_gate_kernel<<<grid, 256, 0, stream>>>(xh, wgh, bg, mixb, out);
}
